// Sylvester_29703993819354
// MI455X (gfx1250) — matrix operand order repaired and checked by reading, not compiled, not run on hardware
//
#include <hip/hip_runtime.h>
#include <hip/hip_bf16.h>
#include <math.h>

// ---------------------------------------------------------------------------
// Sylvester flow on MI455X (gfx1250, wave32).
//   z:(65536,512) v:(511,512) Rs:(512,512) r2diag:(512,) c:(1,512)
//   out = [z_out (65536*512 f32) | ldj (65536 f32)]
// Big GEMMs (2 x 65536x512x512 = 69 GFLOP) via v_wmma_f32_16x16x32_f16 with
// f32 accumulate.  B panels staged in LDS (conflict-free padded stride),
// all 4 b-fragments loaded ahead of the WMMA chain, A stream double-buffered
// with k-unroll-by-2 (no register-rotate movs), global_prefetch_b8 ahead.
// ---------------------------------------------------------------------------

typedef _Float16 v16h __attribute__((ext_vector_type(16)));
typedef _Float16 v8h  __attribute__((ext_vector_type(8)));
typedef float    v8f  __attribute__((ext_vector_type(8)));

#define ZD 512
#define MD 512
#define HH 511
#define BATCH 65536

// LDS B-panel row stride in halves: 512 + 8 pad => 260 dwords per row,
// 260 % 64 == 4, so the 16 lanes of a b128 fragment load (4 dwords each)
// cover all 64 LDS banks exactly once -> conflict-free ds_load_b128.
#define BSTRIDE 520

// ---------------------------------------------------------------------------
// Kernel 1: normalize Householder vectors.  one block per row (511 blocks).
// ---------------------------------------------------------------------------
__global__ void prep_vn_kernel(const float* __restrict__ v, float* __restrict__ vn) {
    const int row = blockIdx.x;
    const int tid = threadIdx.x;           // 256 threads
    float a0 = v[row * ZD + tid];
    float a1 = v[row * ZD + tid + 256];
    float p = a0 * a0 + a1 * a1;
    #pragma unroll
    for (int off = 16; off > 0; off >>= 1) p += __shfl_xor(p, off, 32);
    __shared__ float red[8];
    __shared__ float nrm;
    if ((tid & 31) == 0) red[tid >> 5] = p;
    __syncthreads();
    if (tid == 0) {
        float s = 0.f;
        #pragma unroll
        for (int w = 0; w < 8; ++w) s += red[w];
        nrm = sqrtf(s);
    }
    __syncthreads();
    const float inv = 1.0f / nrm;
    vn[row * ZD + tid]       = a0 * inv;
    vn[row * ZD + tid + 256] = a1 * inv;
}

// ---------------------------------------------------------------------------
// Kernel 2: q = prod_i (I - 2 v_i v_i^T).  Row-independent:
//   q_row <- q_row - 2 (q_row . v_i) v_i      for i = 0..510 (in order)
// One block per row of q (512 blocks), q_row lives in LDS.
// ---------------------------------------------------------------------------
__global__ void build_q_kernel(const float* __restrict__ vn, float* __restrict__ q) {
    __shared__ float qrow[ZD];
    __shared__ float red[8];
    __shared__ float dotS;
    const int row = blockIdx.x;
    const int tid = threadIdx.x;           // 256 threads
    qrow[tid]       = (tid == row) ? 1.f : 0.f;
    qrow[tid + 256] = (tid + 256 == row) ? 1.f : 0.f;
    __syncthreads();
    for (int i = 0; i < HH; ++i) {
        const float* vi = vn + i * ZD;
        float v0 = vi[tid], v1 = vi[tid + 256];
        float p = qrow[tid] * v0 + qrow[tid + 256] * v1;
        #pragma unroll
        for (int off = 16; off > 0; off >>= 1) p += __shfl_xor(p, off, 32);
        if ((tid & 31) == 0) red[tid >> 5] = p;
        __syncthreads();
        if (tid == 0) {
            float d = 0.f;
            #pragma unroll
            for (int w = 0; w < 8; ++w) d += red[w];
            dotS = d;
        }
        __syncthreads();
        const float d2 = 2.0f * dotS;
        qrow[tid]       -= d2 * v0;
        qrow[tid + 256] -= d2 * v1;
        __syncthreads();
    }
    q[row * ZD + tid]       = qrow[tid];
    q[row * ZD + tid + 256] = qrow[tid + 256];
}

// ---------------------------------------------------------------------------
// Kernel 3: form WMMA operand matrices (f16) + rrii.
//   BtT[m][i] = Bt[i][m]   (K(=z-dim i)-contiguous for GEMM1 B-fragments)
//   Af [i][m] = A[i][m]    (row-major => K(=m)-contiguous for GEMM2, B=A^T)
//   rrii[m]   = Rs[m][m]*r2diag[m]
// ---------------------------------------------------------------------------
__global__ void prep_mats_kernel(const float* __restrict__ q,
                                 const float* __restrict__ Rs,
                                 const float* __restrict__ r2d,
                                 _Float16* __restrict__ BtT,
                                 _Float16* __restrict__ Af,
                                 float* __restrict__ rrii) {
    const int g = blockIdx.x * 256 + threadIdx.x;   // 262144 threads
    const int i = g >> 9;
    const int m = g & 511;
    const float* qi  = q + i * ZD;
    const float* rsm = Rs + m * MD;
    float bt = qi[m] * r2d[m];
    for (int k = m + 1; k < MD; ++k) bt += qi[k] * rsm[k];
    float a = 0.f;
    for (int k = 0; k <= m; ++k) a += qi[k] * Rs[k * MD + m];
    BtT[m * ZD + i] = (_Float16)bt;
    Af [i * MD + m] = (_Float16)a;
    if (i == 0) rrii[m] = Rs[m * MD + m] * r2d[m];
}

// ---------------------------------------------------------------------------
// WMMA fragment helpers (CDNA5 16x16x32 f16 layouts, wave32).
// A (16xK=32): lane&15 = row M; lanes 0-15 hold K {0..7,16..23},
//              lanes 16-31 hold K {8..15,24..31}  (VGPRs 0-3 / 4-7).
// B (K=32x16): lane&15 = col N; lanes 0-15 hold K {0..7,16..23}, lanes 16-31 K {8..15,24..31}.
// ---------------------------------------------------------------------------
__device__ __forceinline__ v16h load_a_f32(const float* __restrict__ base,
                                           int row, int k0, int lane) {
    const int kb = (lane < 16) ? 0 : 8;
    const float* p = base + (size_t)row * ZD + k0 + kb;
    v8f lo = *reinterpret_cast<const v8f*>(p);
    v8f hi = *reinterpret_cast<const v8f*>(p + 16);
    v8h l2 = __builtin_convertvector(lo, v8h);
    v8h h2 = __builtin_convertvector(hi, v8h);
    return __builtin_shufflevector(l2, h2, 0,1,2,3,4,5,6,7,8,9,10,11,12,13,14,15);
}

__device__ __forceinline__ v16h load_a_f16(const _Float16* __restrict__ base,
                                           int row, int k0, int lane) {
    const int kb = (lane < 16) ? 0 : 8;
    const _Float16* p = base + (size_t)row * MD + k0 + kb;
    v8h lo = *reinterpret_cast<const v8h*>(p);
    v8h hi = *reinterpret_cast<const v8h*>(p + 16);
    return __builtin_shufflevector(lo, hi, 0,1,2,3,4,5,6,7,8,9,10,11,12,13,14,15);
}

// b-fragment from LDS panel (row-padded to BSTRIDE halves, 16B aligned)
__device__ __forceinline__ v16h load_b_lds(const _Float16* bs, int nloc,
                                           int k0, int lane) {
    const int kb = (lane < 16) ? 0 : 8;
    const _Float16* p = bs + nloc * BSTRIDE + k0 + kb;
    v8h lo = *reinterpret_cast<const v8h*>(p);
    v8h hi = *reinterpret_cast<const v8h*>(p + 16);
    return __builtin_shufflevector(lo, hi, 0,1,2,3,4,5,6,7,8,9,10,11,12,13,14,15);
}

// cooperative global->LDS copy of a 64 x 512 f16 panel (rows col0..col0+63)
__device__ __forceinline__ void stage_b_panel(_Float16* bs,
                                              const _Float16* __restrict__ src,
                                              int col0, int tid) {
    #pragma unroll
    for (int it = 0; it < 16; ++it) {
        const int chunk = it * 256 + tid;      // 4096 chunks of 8 halves
        const int row   = chunk >> 6;          // 64 chunks per row
        const int off   = (chunk & 63) * 8;
        *reinterpret_cast<v8h*>(bs + row * BSTRIDE + off) =
            *reinterpret_cast<const v8h*>(src + (size_t)(col0 + row) * 512 + off);
    }
}

// issue all 4 b-fragment loads, then the 4 WMMAs (lets the compiler group the
// 8 ds_load_b128 and overlap them with the matrix pipe)
__device__ __forceinline__ void wmma_step(v8f acc[4], const v16h& a,
                                          const _Float16* bs, int nl,
                                          int k0, int lane) {
    v16h b0 = load_b_lds(bs,      nl, k0, lane);
    v16h b1 = load_b_lds(bs, 16 + nl, k0, lane);
    v16h b2 = load_b_lds(bs, 32 + nl, k0, lane);
    v16h b3 = load_b_lds(bs, 48 + nl, k0, lane);
    acc[0] = __builtin_amdgcn_wmma_f32_16x16x32_f16(false, a, false, b0,
                                                    (short)0, acc[0], false, false);
    acc[1] = __builtin_amdgcn_wmma_f32_16x16x32_f16(false, a, false, b1,
                                                    (short)0, acc[1], false, false);
    acc[2] = __builtin_amdgcn_wmma_f32_16x16x32_f16(false, a, false, b2,
                                                    (short)0, acc[2], false, false);
    acc[3] = __builtin_amdgcn_wmma_f32_16x16x32_f16(false, a, false, b3,
                                                    (short)0, acc[3], false, false);
}

// ---------------------------------------------------------------------------
// Kernel 4: hB = tanh(z @ Bt + c), stored f16.
// Block = 256 thr (8 waves); block tile 128(batch) x 64(m); wave tile 16x64.
// B panel in LDS; A double-buffered, k unrolled by 2 (no rotate movs).
// ---------------------------------------------------------------------------
__global__ void __launch_bounds__(256)
gemm_tanh_kernel(const float* __restrict__ z,
                 const _Float16* __restrict__ BtT,
                 const float* __restrict__ c,
                 _Float16* __restrict__ hB) {
    __shared__ __align__(16) _Float16 bs[64 * BSTRIDE];
    const int tid  = threadIdx.x;
    const int lane = tid & 31;
    const int nl   = lane & 15;
    const int wave = tid >> 5;
    const int r0   = blockIdx.x * 128 + wave * 16;
    const int col0 = blockIdx.y * 64;
    const int arow = r0 + nl;

    stage_b_panel(bs, BtT, col0, tid);
    __syncthreads();

    v8f acc[4];
    #pragma unroll
    for (int t = 0; t < 4; ++t) acc[t] = (v8f){0.f,0.f,0.f,0.f,0.f,0.f,0.f,0.f};

    v16h a0 = load_a_f32(z, arow, 0, lane);
    #pragma unroll 2
    for (int k0 = 0; k0 < ZD; k0 += 64) {
        v16h a1 = load_a_f32(z, arow, k0 + 32, lane);
        __builtin_prefetch(z + (size_t)arow * ZD + k0 + 128, 0, 1);
        wmma_step(acc, a0, bs, nl, k0, lane);
        if (k0 + 64 < ZD) a0 = load_a_f32(z, arow, k0 + 64, lane);
        wmma_step(acc, a1, bs, nl, k0 + 32, lane);
    }

    const int rbase = r0 + ((lane < 16) ? 0 : 8);
    #pragma unroll
    for (int t = 0; t < 4; ++t) {
        const int n = col0 + t * 16 + nl;
        const float cn = c[n];
        #pragma unroll
        for (int j = 0; j < 8; ++j) {
            float h = tanhf(acc[t][j] + cn);
            hB[(size_t)(rbase + j) * MD + n] = (_Float16)h;
        }
    }
}

// ---------------------------------------------------------------------------
// Kernel 5: z_out = hB @ A^T + z   (B-matrix = A^T, K-major = Af row-major).
// ---------------------------------------------------------------------------
__global__ void __launch_bounds__(256)
gemm_zout_kernel(const _Float16* __restrict__ hB,
                 const _Float16* __restrict__ Af,
                 const float* __restrict__ z,
                 float* __restrict__ zout) {
    __shared__ __align__(16) _Float16 bs[64 * BSTRIDE];
    const int tid  = threadIdx.x;
    const int lane = tid & 31;
    const int nl   = lane & 15;
    const int wave = tid >> 5;
    const int r0   = blockIdx.x * 128 + wave * 16;
    const int col0 = blockIdx.y * 64;
    const int arow = r0 + nl;

    stage_b_panel(bs, Af, col0, tid);   // Af[n][k] contiguous in k
    __syncthreads();

    v8f acc[4];
    #pragma unroll
    for (int t = 0; t < 4; ++t) acc[t] = (v8f){0.f,0.f,0.f,0.f,0.f,0.f,0.f,0.f};

    v16h a0 = load_a_f16(hB, arow, 0, lane);
    #pragma unroll 2
    for (int k0 = 0; k0 < MD; k0 += 64) {
        v16h a1 = load_a_f16(hB, arow, k0 + 32, lane);
        __builtin_prefetch(hB + (size_t)arow * MD + k0 + 128, 0, 1);
        wmma_step(acc, a0, bs, nl, k0, lane);
        if (k0 + 64 < MD) a0 = load_a_f16(hB, arow, k0 + 64, lane);
        wmma_step(acc, a1, bs, nl, k0 + 32, lane);
    }

    const int rbase = r0 + ((lane < 16) ? 0 : 8);
    #pragma unroll
    for (int t = 0; t < 4; ++t) {
        const int n = col0 + t * 16 + nl;
        #pragma unroll
        for (int j = 0; j < 8; ++j) {
            const size_t idx = (size_t)(rbase + j) * ZD + n;
            zout[idx] = acc[t][j] + z[idx];
        }
    }
}

// ---------------------------------------------------------------------------
// Kernel 6: ldj[b] = sum_m log|(1 - hB^2) * rrii[m] + 1|.  One wave per row.
// ---------------------------------------------------------------------------
__global__ void ldj_kernel(const _Float16* __restrict__ hB,
                           const float* __restrict__ rrii,
                           float* __restrict__ ldj) {
    const int lane = threadIdx.x & 31;
    const int wave = threadIdx.x >> 5;
    const int row  = blockIdx.x * 8 + wave;
    const _Float16* hrow = hB + (size_t)row * MD;
    float s = 0.f;
    #pragma unroll
    for (int j = 0; j < 16; ++j) {
        const int m = j * 32 + lane;
        const float h = (float)hrow[m];
        const float der = 1.f - h * h;
        s += logf(fabsf(der * rrii[m] + 1.0f));
    }
    #pragma unroll
    for (int off = 16; off > 0; off >>= 1) s += __shfl_xor(s, off, 32);
    if (lane == 0) ldj[row] = s;
}

// ---------------------------------------------------------------------------
extern "C" void kernel_launch(void* const* d_in, const int* in_sizes, int n_in,
                              void* d_out, int out_size, void* d_ws, size_t ws_size,
                              hipStream_t stream) {
    (void)in_sizes; (void)n_in; (void)out_size; (void)ws_size;
    const float* z   = (const float*)d_in[0];
    const float* v   = (const float*)d_in[1];
    const float* Rs  = (const float*)d_in[2];
    const float* r2d = (const float*)d_in[3];
    const float* c   = (const float*)d_in[4];

    char* ws = (char*)d_ws;
    float*    vn   = (float*)   (ws + 0);                      // ~1 MB
    float*    q    = (float*)   (ws + (1u << 20));             // 1 MB
    _Float16* BtT  = (_Float16*)(ws + (2u << 20));             // 512 KB
    _Float16* Af   = (_Float16*)(ws + (2u << 20) + 524288u);   // 512 KB
    float*    rrii = (float*)   (ws + (3u << 20));             // 2 KB
    _Float16* hB   = (_Float16*)(ws + (4u << 20));             // 64 MB

    float* zout = (float*)d_out;
    float* ldj  = zout + (size_t)BATCH * ZD;

    prep_vn_kernel  <<<HH,  256, 0, stream>>>(v, vn);
    build_q_kernel  <<<ZD,  256, 0, stream>>>(vn, q);
    prep_mats_kernel<<<1024, 256, 0, stream>>>(q, Rs, r2d, BtT, Af, rrii);
    gemm_tanh_kernel<<<dim3(BATCH / 128, MD / 64), 256, 0, stream>>>(z, BtT, c, hB);
    gemm_zout_kernel<<<dim3(BATCH / 128, ZD / 64), 256, 0, stream>>>(hB, Af, z, zout);
    ldj_kernel      <<<BATCH / 8, 256, 0, stream>>>(hB, rrii, ldj);
}
